// GCNCombiner_69612829933735
// MI455X (gfx1250) — compile-verified, matrix operand order checked by reading, not run on hardware
//
#include <hip/hip_runtime.h>
#include <hip/hip_bf16.h>
#include <math.h>
#include <stdint.h>

typedef __attribute__((ext_vector_type(16))) __bf16 bf16x16;
typedef __attribute__((ext_vector_type(8)))  __bf16 bf16x8;
typedef __attribute__((ext_vector_type(8)))  float  f32x8;
typedef __attribute__((ext_vector_type(4)))  float  f32x4;

#define TILE_M 128
#define TILE_N 128
#define TILE_K 32
#define LDSP   40   // padded LDS pitch (elements); 80B rows keep 16B alignment, skew banks

// ---------------------------------------------------------------------------
// WMMA fragment gather from an LDS row laid out [row][k0..k31] (bf16).
// Per ISA 7.12.2 (16-bit A 16x32): lane L holds row M=L%16; its 16 elements cover
// K = {kb..kb+7} U {kb+16..kb+23} with kb = 8*(L/16). Two aligned 16B LDS loads.
// ---------------------------------------------------------------------------
__device__ __forceinline__ bf16x16 frag_from_lds(const __bf16* rowp, int kb) {
  bf16x8 lo = *(const bf16x8*)(rowp + kb);
  bf16x8 hi = *(const bf16x8*)(rowp + kb + 16);
  bf16x16 r;
#pragma unroll
  for (int i = 0; i < 8; ++i) { r[i] = lo[i]; r[i + 8] = hi[i]; }
  return r;
}

// 16-byte async DMA: global -> LDS, tracked by ASYNCcnt (ISA §10 / 15.18.3 op 98).
__device__ __forceinline__ void async_copy16(unsigned lds_off, const void* gptr) {
  asm volatile("global_load_async_to_lds_b128 %0, %1, off"
               :: "v"(lds_off), "v"((unsigned long long)(uintptr_t)gptr)
               : "memory");
}

// ---------------------------------------------------------------------------
// Async double-buffered WMMA GEMM (bf16 inputs):
//   C[m,n] = act( sum_k A[m,k]*B[n,k] + biasN[n] )        (A@B^T)
// Requirements: M % 128 == 0, N % 128 == 0, K % 32 == 0 (true for all projections).
// Output row m is remapped through (m/Mseg, m%Mseg, strideCseg) so level outputs
// scatter directly into the concatenated hs[B,T,P].
// ---------------------------------------------------------------------------
template<typename OUT, bool RELU>
__global__ __launch_bounds__(256)
void gemm_wmma_async(const __bf16* __restrict__ A, int lda,
                     const __bf16* __restrict__ Bm, int ldb,
                     const float* __restrict__ biasN,
                     OUT* __restrict__ C, int ldc,
                     int M, int N, int K, int Mseg, long long strideCseg)
{
  __shared__ __bf16 sA[2][TILE_M * LDSP];
  __shared__ __bf16 sB[2][TILE_N * LDSP];

  const int tid  = threadIdx.x;
  const int lane = tid & 31;
  const int wv   = tid >> 5;
  const int wm   = (wv >> 2) * 64;
  const int wn   = (wv & 3) * 32;
  const int l16  = lane & 15;
  const int lhi  = lane >> 4;
  const int kb   = lhi * 8;
  const int bm   = blockIdx.x * TILE_M;
  const int bn   = blockIdx.y * TILE_N;

  // LDS byte offsets (flat shared pointer truncated to 32 bits = offset in wave LDS)
  const unsigned aoff[2] = { (unsigned)(uintptr_t)&sA[0][0], (unsigned)(uintptr_t)&sA[1][0] };
  const unsigned boff[2] = { (unsigned)(uintptr_t)&sB[0][0], (unsigned)(uintptr_t)&sB[1][0] };

  const __bf16* Abase = A  + (long long)bm * lda;
  const __bf16* Bbase = Bm + (long long)bn * ldb;

  // each thread DMAs 2x16B of A and 2x16B of B per tile: chunk c -> row c>>2, k-off (c&3)*8
  const int r0 = tid >> 2,          ko0 = (tid & 3) * 8;          // chunk tid
  const int r1 = (tid + 256) >> 2,  ko1 = ((tid + 256) & 3) * 8;  // chunk tid+256

#define ISSUE_TILE(buf, kofs)                                                        \
  do {                                                                               \
    async_copy16(aoff[buf] + (unsigned)((r0 * LDSP + ko0) * 2),                      \
                 Abase + (long long)r0 * lda + (kofs) + ko0);                        \
    async_copy16(boff[buf] + (unsigned)((r0 * LDSP + ko0) * 2),                      \
                 Bbase + (long long)r0 * ldb + (kofs) + ko0);                        \
    async_copy16(aoff[buf] + (unsigned)((r1 * LDSP + ko1) * 2),                      \
                 Abase + (long long)r1 * lda + (kofs) + ko1);                        \
    async_copy16(boff[buf] + (unsigned)((r1 * LDSP + ko1) * 2),                      \
                 Bbase + (long long)r1 * ldb + (kofs) + ko1);                        \
  } while (0)

  f32x8 acc[4][2];
#pragma unroll
  for (int i = 0; i < 4; ++i)
#pragma unroll
    for (int j = 0; j < 2; ++j)
#pragma unroll
      for (int r = 0; r < 8; ++r) acc[i][j][r] = 0.0f;

  ISSUE_TILE(0, 0);                      // prefetch tile 0 into buffer 0
  const int KT = K / TILE_K;
  for (int kt = 0; kt < KT; ++kt) {
    const int cur = kt & 1;
    if (kt + 1 < KT) {
      ISSUE_TILE(cur ^ 1, (kt + 1) * TILE_K);   // buffer cur^1 was consumed in kt-1
      asm volatile("s_wait_asynccnt 0x4" ::: "memory");  // current tile's 4 DMAs done
    } else {
      asm volatile("s_wait_asynccnt 0x0" ::: "memory");
    }
    __syncthreads();

    const __bf16* cA = sA[cur];
    const __bf16* cB = sB[cur];
    bf16x16 af[4], bfr[2];
#pragma unroll
    for (int i = 0; i < 4; ++i)
      af[i] = frag_from_lds(&cA[(wm + i * 16 + l16) * LDSP], kb);
#pragma unroll
    for (int j = 0; j < 2; ++j)
      bfr[j] = frag_from_lds(&cB[(wn + j * 16 + l16) * LDSP], kb);
#pragma unroll
    for (int i = 0; i < 4; ++i)
#pragma unroll
      for (int j = 0; j < 2; ++j)
        acc[i][j] = __builtin_amdgcn_wmma_f32_16x16x32_bf16(
            false, af[i], false, bfr[j], (short)0, acc[i][j], false, false);
    __syncthreads();
  }
#undef ISSUE_TILE

  // epilogue: ISA C/D layout -> row = base + r + 8*(lane/16), col = lane%16
#pragma unroll
  for (int i = 0; i < 4; ++i) {
#pragma unroll
    for (int j = 0; j < 2; ++j) {
      const int col = bn + wn + j * 16 + l16;
#pragma unroll
      for (int r = 0; r < 8; ++r) {
        const int row = bm + wm + i * 16 + lhi * 8 + r;
        float v = acc[i][j][r];
        if (biasN) v += biasN[col];
        if (RELU)  v = fmaxf(v, 0.0f);
        const int bseg = row / Mseg;
        const int rs   = row - bseg * Mseg;
        C[(long long)bseg * strideCseg + (long long)rs * ldc + col] = (OUT)v;
      }
    }
  }
}

// ---------------------------------------------------------------------------
// Guarded f32-input WMMA GEMM (VALU-staged f32->bf16) for the small batched
// GEMMs that need transposed staging (pool0: B stored [K,N]) or M=170 guards.
// ---------------------------------------------------------------------------
template<bool TRANSB, bool RELU>
__global__ __launch_bounds__(256)
void gemm_wmma_bf16(const float* __restrict__ A, long long strideA, int lda,
                    const float* __restrict__ Bm, long long strideB, int ldb,
                    const float* __restrict__ biasN, const float* __restrict__ biasM,
                    float* __restrict__ C, long long strideC, int ldc,
                    int M, int N, int K, int Mseg, long long strideCseg)
{
  __shared__ __bf16 sA[TILE_M * LDSP];
  __shared__ __bf16 sB[TILE_N * LDSP];

  const int tid  = threadIdx.x;
  const int lane = tid & 31;
  const int wv   = tid >> 5;
  const int wm   = (wv >> 2) * 64;
  const int wn   = (wv & 3) * 32;
  const int l16  = lane & 15;
  const int lhi  = lane >> 4;
  const int kb   = lhi * 8;

  const int bm = blockIdx.x * TILE_M;
  const int bn = blockIdx.y * TILE_N;
  const int z  = blockIdx.z;
  A  += (long long)z * strideA;
  Bm += (long long)z * strideB;

  f32x8 acc[4][2];
#pragma unroll
  for (int i = 0; i < 4; ++i)
#pragma unroll
    for (int j = 0; j < 2; ++j)
#pragma unroll
      for (int r = 0; r < 8; ++r) acc[i][j][r] = 0.0f;

  const int ar  = tid >> 1;
  const int ac0 = (tid & 1) * 16;
  const int bkk = tid >> 3;
  const int bc0 = (tid & 7) * 16;

  for (int k0 = 0; k0 < K; k0 += TILE_K) {
    {
      const int gm = bm + ar;
      __bf16* dst = &sA[ar * LDSP + ac0];
      if (gm < M) {
        const float* src = A + (long long)gm * lda + k0 + ac0;
        if (k0 + TILE_K < K) __builtin_prefetch(src + TILE_K, 0, 1);
#pragma unroll
        for (int v = 0; v < 4; ++v) {
          f32x4 f = *(const f32x4*)(src + v * 4);
          dst[v*4+0] = (__bf16)f.x; dst[v*4+1] = (__bf16)f.y;
          dst[v*4+2] = (__bf16)f.z; dst[v*4+3] = (__bf16)f.w;
        }
      } else {
#pragma unroll
        for (int v = 0; v < 16; ++v) dst[v] = (__bf16)0.0f;
      }
    }
    if (TRANSB) {
      const int gn = bn + ar;
      __bf16* dst = &sB[ar * LDSP + ac0];
      if (gn < N) {
        const float* src = Bm + (long long)gn * ldb + k0 + ac0;
#pragma unroll
        for (int v = 0; v < 4; ++v) {
          f32x4 f = *(const f32x4*)(src + v * 4);
          dst[v*4+0] = (__bf16)f.x; dst[v*4+1] = (__bf16)f.y;
          dst[v*4+2] = (__bf16)f.z; dst[v*4+3] = (__bf16)f.w;
        }
      } else {
#pragma unroll
        for (int v = 0; v < 16; ++v) dst[v] = (__bf16)0.0f;
      }
    } else {
      const float* src = Bm + (long long)(k0 + bkk) * ldb + bn + bc0;
#pragma unroll
      for (int v = 0; v < 16; ++v) {
        const int n = bn + bc0 + v;
        const float f = (n < N) ? src[v] : 0.0f;
        sB[(bc0 + v) * LDSP + bkk] = (__bf16)f;
      }
    }
    __syncthreads();

    bf16x16 af[4], bfr[2];
#pragma unroll
    for (int i = 0; i < 4; ++i)
      af[i] = frag_from_lds(&sA[(wm + i * 16 + l16) * LDSP], kb);
#pragma unroll
    for (int j = 0; j < 2; ++j)
      bfr[j] = frag_from_lds(&sB[(wn + j * 16 + l16) * LDSP], kb);
#pragma unroll
    for (int i = 0; i < 4; ++i)
#pragma unroll
      for (int j = 0; j < 2; ++j)
        acc[i][j] = __builtin_amdgcn_wmma_f32_16x16x32_bf16(
            false, af[i], false, bfr[j], (short)0, acc[i][j], false, false);
    __syncthreads();
  }

  C += (long long)z * strideC;
#pragma unroll
  for (int i = 0; i < 4; ++i) {
#pragma unroll
    for (int j = 0; j < 2; ++j) {
      const int col = bn + wn + j * 16 + l16;
#pragma unroll
      for (int r = 0; r < 8; ++r) {
        const int row = bm + wm + i * 16 + lhi * 8 + r;
        if (row < M && col < N) {
          float v = acc[i][j][r];
          if (biasN) v += biasN[col];
          if (biasM) v += biasM[row];
          if (RELU)  v = fmaxf(v, 0.0f);
          const int bseg = row / Mseg;
          const int rs   = row - bseg * Mseg;
          C[(long long)bseg * strideCseg + (long long)rs * ldc + col] = v;
        }
      }
    }
  }
}

// ---- small kernels ---------------------------------------------------------

__global__ __launch_bounds__(256)
void k_cvt(const float* __restrict__ src, __bf16* __restrict__ dst, long long n) {
  long long i = ((long long)blockIdx.x * 256 + threadIdx.x) * 4;
  if (i + 3 < n) {
    f32x4 f = *(const f32x4*)(src + i);
    dst[i]   = (__bf16)f.x; dst[i+1] = (__bf16)f.y;
    dst[i+2] = (__bf16)f.z; dst[i+3] = (__bf16)f.w;
  } else {
    for (long long k = i; k < n; ++k) dst[k] = (__bf16)src[k];
  }
}

__global__ __launch_bounds__(256)
void k_qkbar(const float* __restrict__ cqw, const float* __restrict__ cqb,
             const float* __restrict__ ckw, const float* __restrict__ ckb,
             float* __restrict__ wqb, float* __restrict__ wkb) {
  for (int p = threadIdx.x; p < 512; p += 256) {
    float sq = 0.f, sk = 0.f;
    for (int q = 0; q < 128; ++q) { sq += cqw[q * 512 + p]; sk += ckw[q * 512 + p]; }
    wqb[p] = sq * (1.f / 128.f);
    wkb[p] = sk * (1.f / 128.f);
  }
  if (threadIdx.x == 0) {
    float s = 0.f, t = 0.f;
    for (int q = 0; q < 128; ++q) { s += cqb[q]; t += ckb[q]; }
    wqb[512] = s * (1.f / 128.f);
    wkb[512] = t * (1.f / 128.f);
  }
}

__global__ __launch_bounds__(256)
void k_qk(const float* __restrict__ hsp, const float* __restrict__ wqb,
          const float* __restrict__ wkb, float* __restrict__ q1,
          float* __restrict__ k1, int BJ) {
  const int idx  = blockIdx.x * 8 + (threadIdx.x >> 5);
  const int lane = threadIdx.x & 31;
  if (idx >= BJ) return;
  const float* row = hsp + (long long)idx * 512;
  float aq = 0.f, ak = 0.f;
  for (int p = lane; p < 512; p += 32) {
    const float h = row[p];
    aq += h * wqb[p];
    ak += h * wkb[p];
  }
  for (int off = 16; off; off >>= 1) {
    aq += __shfl_down(aq, off, 32);
    ak += __shfl_down(ak, off, 32);
  }
  if (lane == 0) { q1[idx] = aq + wqb[512]; k1[idx] = ak + wkb[512]; }
}

__global__ __launch_bounds__(256)
void k_coef(const float* __restrict__ q1, const float* __restrict__ k1,
            const float* __restrict__ adj, const float* __restrict__ alpha,
            const float* __restrict__ pw, float* __restrict__ cbj, int B, int J) {
  const int idx = blockIdx.x * blockDim.x + threadIdx.x;
  if (idx >= B * J) return;
  const int b = idx / J, j = idx - b * J;
  const float qv = q1[b * J + j], al = alpha[0];
  float acc = 0.f;
  for (int k = 0; k < J; ++k)
    acc += pw[k] * (adj[j * J + k] + al * tanhf(qv - k1[b * J + k]));
  cbj[idx] = acc;
}

__global__ __launch_bounds__(256)
void k_pool(const float* __restrict__ hc, const float* __restrict__ cbj,
            const float* __restrict__ bng, const float* __restrict__ bnb,
            const float* __restrict__ bnm, const float* __restrict__ bnv,
            const float* __restrict__ pw, const float* __restrict__ pb,
            float* __restrict__ g, int J) {
  const int b = blockIdx.x;
  const int p = blockIdx.y * 256 + threadIdx.x;
  const float* hb = hc + (long long)b * J * 512;
  const float* cb = cbj + b * J;
  float s = 0.f;
  for (int j = 0; j < J; ++j) s += cb[j] * hb[j * 512 + p];
  float W = 0.f;
  for (int k = 0; k < J; ++k) W += pw[k];
  const float sc = bng[p] * rsqrtf(bnv[p] + 1e-5f);
  g[b * 512 + p] = sc * s + (bnb[p] - bnm[p] * sc) * W + pb[0];
}

__global__ __launch_bounds__(256)
void k_cls(const float* __restrict__ g, const float* __restrict__ w,
           const float* __restrict__ bias, float* __restrict__ out, int NC) {
  __shared__ float gs[512];
  const int b = blockIdx.x;
  for (int p = threadIdx.x; p < 512; p += 256) gs[p] = g[b * 512 + p];
  __syncthreads();
  const int n = threadIdx.x;
  if (n < NC) {
    float acc = bias[n];
    const float* wr = w + (long long)n * 512;
    for (int p = 0; p < 512; ++p) acc += gs[p] * wr[p];
    out[b * NC + n] = acc;
  }
}

extern "C" void kernel_launch(void* const* d_in, const int* in_sizes, int n_in,
                              void* d_out, int out_size, void* d_ws, size_t ws_size,
                              hipStream_t stream) {
  const int B = 64, P = 512, T = 2720, NC = 200, J = 170;
  const int SS[4] = {2048, 512, 128, 32};
  const int CS[4] = {256, 512, 1024, 2048};
  (void)in_sizes; (void)n_in; (void)out_size; (void)ws_size;

  // ---- workspace carve-out (64B aligned) ----
  char* base = (char*)d_ws;
  size_t cur = 0;
  auto alloc = [&](size_t bytes) -> char* {
    char* p = base + cur;
    cur += (bytes + 63) & ~(size_t)63;
    return p;
  };
  float*  hs   = (float*)alloc((size_t)B * T * P * 4);      // [B,T,P] f32
  float*  hsp  = (float*)alloc((size_t)B * J * P * 4);      // [B,J,P] f32
  float*  hc   = (float*)alloc((size_t)B * J * P * 4);      // [B,J,P] f32
  float*  q1   = (float*)alloc((size_t)B * J * 4);
  float*  k1   = (float*)alloc((size_t)B * J * 4);
  float*  wqb  = (float*)alloc(513 * 4);
  float*  wkb  = (float*)alloc(513 * 4);
  float*  cbj  = (float*)alloc((size_t)B * J * 4);
  float*  g    = (float*)alloc((size_t)B * P * 4);
  __bf16* hidb = (__bf16*)alloc((size_t)B * 2048 * P * 2);  // bf16 hidden (max level)
  __bf16* xb[4];
  __bf16* w1b[4];
  __bf16* w2b[4];
  for (int i = 0; i < 4; ++i) {
    xb[i]  = (__bf16*)alloc((size_t)B * SS[i] * CS[i] * 2);
    w1b[i] = (__bf16*)alloc((size_t)P * CS[i] * 2);
    w2b[i] = (__bf16*)alloc((size_t)P * P * 2);
  }

  dim3 blk(256);
  auto cvt = [&](const void* src, __bf16* dst, long long n) {
    k_cvt<<<dim3((unsigned)((n + 1023) / 1024)), blk, 0, stream>>>((const float*)src, dst, n);
  };

  // ---- pre-convert activations + projection weights to bf16 ----
  for (int i = 0; i < 4; ++i) {
    cvt(d_in[i * 5 + 0], xb[i],  (long long)B * SS[i] * CS[i]);
    cvt(d_in[i * 5 + 1], w1b[i], (long long)P * CS[i]);
    cvt(d_in[i * 5 + 3], w2b[i], (long long)P * P);
  }

  // ---- four projection levels via async double-buffered WMMA GEMM ----
  int t0 = 0;
  for (int i = 0; i < 4; ++i) {
    const float* b1 = (const float*)d_in[i * 5 + 2];
    const float* b2 = (const float*)d_in[i * 5 + 4];
    const int S = SS[i], C = CS[i], M = B * S;
    dim3 grd(M / TILE_M, P / TILE_N, 1);
    gemm_wmma_async<__bf16, true><<<grd, blk, 0, stream>>>(
        xb[i], C, w1b[i], C, b1, hidb, P, M, P, C, M, 0);
    gemm_wmma_async<float, false><<<grd, blk, 0, stream>>>(
        hidb, P, w2b[i], P, b2, hs + (long long)t0 * P, P, M, P, P, S, (long long)T * P);
    t0 += S;
  }

  // ---- pool0 (batched, B stored [K,N]): hsp[b] = pool0_w @ hs[b] + pool0_b[j] ----
  {
    dim3 grd((J + TILE_M - 1) / TILE_M, P / TILE_N, B);
    gemm_wmma_bf16<false, false><<<grd, blk, 0, stream>>>(
        (const float*)d_in[20], 0, T,
        hs, (long long)T * P, P,
        nullptr, (const float*)d_in[21],
        hsp, (long long)J * P, P, J, P, T, J, 0);
  }

  // ---- collapsed attention coefficients ----
  k_qkbar<<<1, blk, 0, stream>>>((const float*)d_in[25], (const float*)d_in[26],
                                 (const float*)d_in[27], (const float*)d_in[28], wqb, wkb);
  k_qk<<<(B * J + 7) / 8, blk, 0, stream>>>(hsp, wqb, wkb, q1, k1, B * J);
  k_coef<<<(B * J + 255) / 256, blk, 0, stream>>>(
      q1, k1, (const float*)d_in[22], (const float*)d_in[29],
      (const float*)d_in[34], cbj, B, J);

  // ---- conv1 (batched): hc[b] = hsp[b] @ conv1_w^T + conv1_b ----
  {
    dim3 grd((J + TILE_M - 1) / TILE_M, P / TILE_N, B);
    gemm_wmma_bf16<true, false><<<grd, blk, 0, stream>>>(
        hsp, (long long)J * P, P,
        (const float*)d_in[23], 0, P,
        (const float*)d_in[24], nullptr,
        hc, (long long)J * P, P, J, P, P, J, 0);
  }

  // ---- fused A1-matmul + BN + pool1 (linear collapse), then classifier ----
  k_pool<<<dim3(B, P / 256), blk, 0, stream>>>(
      hc, cbj, (const float*)d_in[30], (const float*)d_in[31],
      (const float*)d_in[32], (const float*)d_in[33],
      (const float*)d_in[34], (const float*)d_in[35], g, J);
  k_cls<<<B, blk, 0, stream>>>(g, (const float*)d_in[36],
                               (const float*)d_in[37], (float*)d_out, NC);
}
